// BlockRNN_78056735638165
// MI455X (gfx1250) — matrix operand order repaired and checked by reading, not compiled, not run on hardware
//
#include <hip/hip_runtime.h>

#define B_    2048
#define E_    256
#define H_    128
#define NH_   8
#define HD_   32
#define G3_   384      // 3*H
#define TOK_  129      // 1 + H tokens per batch row
#define EPS_  1e-5f
#define WTILE_ (16 * G3_)   // uint4 entries of one channel's Whh image (98304 B)

typedef float  v8f   __attribute__((ext_vector_type(8)));
typedef __bf16 v16bf __attribute__((ext_vector_type(16)));
typedef int    v4i_  __attribute__((ext_vector_type(4)));

#if defined(__has_builtin)
#if __has_builtin(__builtin_amdgcn_global_load_async_to_lds_b128) && \
    __has_builtin(__builtin_amdgcn_s_wait_asynccnt)
#define HAVE_ASYNC_LDS 1
#endif
#endif
#ifndef HAVE_ASYNC_LDS
#define HAVE_ASYNC_LDS 0
#endif

__device__ __forceinline__ unsigned short f2bf(float f) {
  unsigned u = __builtin_bit_cast(unsigned, f);
  u += 0x7fffu + ((u >> 16) & 1u);          // round-to-nearest-even
  return (unsigned short)(u >> 16);
}
__device__ __forceinline__ float bf2f(unsigned short h) {
  unsigned u = ((unsigned)h) << 16;
  return __builtin_bit_cast(float, u);
}

// ---------------------------------------------------------------------------
// Kernel 0a: fp32 -> bf16 weight conversion (weights stay L2-resident)
// ---------------------------------------------------------------------------
__global__ void cvt_kernel(const float* __restrict__ src,
                           unsigned short* __restrict__ dst, int n) {
  int i = blockIdx.x * 256 + threadIdx.x;
  if (i < n) dst[i] = f2bf(src[i]);
}

// ---------------------------------------------------------------------------
// Kernel 0b: pre-swizzle Whh into the exact bf16 LDS image the GRU kernel
// wants: entry (k8*384 + g) = 8 bf16 of row g, k-chunk k8. This turns the
// per-WG weight stage into a straight 96 KB async DMA.
// ---------------------------------------------------------------------------
__global__ __launch_bounds__(128) void cvt_whh_kernel(
    const float* __restrict__ src, uint4* __restrict__ dst) {
  const int e = blockIdx.x;
  for (int idx = threadIdx.x; idx < WTILE_; idx += 128) {
    const int k8 = idx / G3_, g = idx % G3_;
    const float* s = src + ((size_t)e * G3_ + g) * H_ + k8 * 8;
    uint4 w;
    w.x = (unsigned)f2bf(s[0]) | ((unsigned)f2bf(s[1]) << 16);
    w.y = (unsigned)f2bf(s[2]) | ((unsigned)f2bf(s[3]) << 16);
    w.z = (unsigned)f2bf(s[4]) | ((unsigned)f2bf(s[5]) << 16);
    w.w = (unsigned)f2bf(s[6]) | ((unsigned)f2bf(s[7]) << 16);
    dst[(size_t)e * WTILE_ + idx] = w;
  }
}

// ---------------------------------------------------------------------------
// Kernel 1: per-channel GRU scan. One workgroup per e (256 WGs, 128 threads).
// Whh[e] lives in LDS as packed bf16 (async-DMA'd); h is fp32 ping-pong in
// LDS; thread j owns hidden index j and its (r,z,n) weight rows -> gating is
// thread-local, one barrier per step.
// ---------------------------------------------------------------------------
__device__ __forceinline__ void mac8(float& acc, const uint4& wv,
                                     const float4& hA, const float4& hB) {
  acc += bf2f((unsigned short)(wv.x & 0xffffu)) * hA.x;
  acc += bf2f((unsigned short)(wv.x >> 16))     * hA.y;
  acc += bf2f((unsigned short)(wv.y & 0xffffu)) * hA.z;
  acc += bf2f((unsigned short)(wv.y >> 16))     * hA.w;
  acc += bf2f((unsigned short)(wv.z & 0xffffu)) * hB.x;
  acc += bf2f((unsigned short)(wv.z >> 16))     * hB.y;
  acc += bf2f((unsigned short)(wv.w & 0xffffu)) * hB.z;
  acc += bf2f((unsigned short)(wv.w >> 16))     * hB.w;
}

__global__ __launch_bounds__(128) void gru_kernel(
    const float* __restrict__ xin,  const float* __restrict__ h0s,
    const float* __restrict__ Wih,  const float* __restrict__ Whh,
    const float* __restrict__ bih,  const float* __restrict__ bhh,
    const uint4* __restrict__ wbhh,
    unsigned short* __restrict__ obuf, float* __restrict__ hfinal) {
  __shared__ uint4 Wl[WTILE_];                      // 98304 B
  __shared__ __align__(16) float hbuf[2][H_];       // 1 KB ping-pong state
  __shared__ float xcol[B_];                        // 8 KB input column

  const int e = blockIdx.x;
  const int j = threadIdx.x;     // 0..127

#if HAVE_ASYNC_LDS
  {  // pure DMA: pre-swizzled bf16 image -> LDS (ASYNCcnt-tracked)
    const uint4* wsrc = wbhh + (size_t)e * WTILE_;
    for (int idx = j; idx < WTILE_; idx += 128) {
      __builtin_amdgcn_global_load_async_to_lds_b128(
          (v4i_*)(wsrc + idx), (v4i_*)(&Wl[idx]), 0, 0);
    }
  }
#else
  (void)wbhh;
  for (int idx = j; idx < WTILE_; idx += 128) {
    const int k8 = idx / G3_, g = idx % G3_;
    const float* s = Whh + ((size_t)e * G3_ + g) * H_ + k8 * 8;
    uint4 w;
    w.x = (unsigned)f2bf(s[0]) | ((unsigned)f2bf(s[1]) << 16);
    w.y = (unsigned)f2bf(s[2]) | ((unsigned)f2bf(s[3]) << 16);
    w.z = (unsigned)f2bf(s[4]) | ((unsigned)f2bf(s[5]) << 16);
    w.w = (unsigned)f2bf(s[6]) | ((unsigned)f2bf(s[7]) << 16);
    Wl[idx] = w;
  }
#endif
  for (int b = j; b < B_; b += 128) xcol[b] = xin[(size_t)b * E_ + e];

  float hj = h0s[e * H_ + j];
  hbuf[0][j] = hj;

  const float wi_r = Wih[e * G3_ + j];
  const float wi_z = Wih[e * G3_ + H_ + j];
  const float wi_n = Wih[e * G3_ + 2 * H_ + j];
  const float bi_r = bih[e * G3_ + j];
  const float bi_z = bih[e * G3_ + H_ + j];
  const float bi_n = bih[e * G3_ + 2 * H_ + j];
  const float bh_r = bhh[e * G3_ + j];
  const float bh_z = bhh[e * G3_ + H_ + j];
  const float bh_n = bhh[e * G3_ + 2 * H_ + j];
#if HAVE_ASYNC_LDS
  __builtin_amdgcn_s_wait_asynccnt(0);
#endif
  __syncthreads();

  #pragma unroll 1
  for (int b = 0; b < B_; ++b) {
    const int cur = b & 1;
    float ar = bh_r, az = bh_z, an = bh_n;
    const float4* hp = (const float4*)hbuf[cur];
    #pragma unroll
    for (int k8 = 0; k8 < 16; ++k8) {
      const float4 hA = hp[2 * k8];
      const float4 hB = hp[2 * k8 + 1];
      const uint4 wr = Wl[k8 * G3_ + j];
      const uint4 wz = Wl[k8 * G3_ + H_ + j];
      const uint4 wn = Wl[k8 * G3_ + 2 * H_ + j];
      mac8(ar, wr, hA, hB);
      mac8(az, wz, hA, hB);
      mac8(an, wn, hA, hB);
    }
    const float x = xcol[b];
    const float r = 1.0f / (1.0f + expf(-(wi_r * x + bi_r + ar)));
    const float z = 1.0f / (1.0f + expf(-(wi_z * x + bi_z + az)));
    const float n = tanhf(wi_n * x + bi_n + r * an);
    hj = (1.0f - z) * n + z * hj;
    hbuf[cur ^ 1][j] = hj;
    obuf[((size_t)b * H_ + j) * E_ + e] = f2bf(hj);
    __syncthreads();
  }
  hfinal[e * H_ + j] = hj;   // (E,H) row-major
}

// ---------------------------------------------------------------------------
// Kernel 2: per-token layernorm stats. token m = b*129 + t.
// ---------------------------------------------------------------------------
__global__ __launch_bounds__(256) void stats_kernel(
    const float* __restrict__ xin, const unsigned short* __restrict__ obuf,
    float2* __restrict__ stats) {
  const int m = blockIdx.x;
  const int b = m / TOK_, t = m % TOK_;
  const int k = threadIdx.x;
  const float v = (t == 0) ? xin[(size_t)b * E_ + k]
                           : bf2f(obuf[((size_t)b * H_ + (t - 1)) * E_ + k]);
  __shared__ float rs[256], rq[256];
  rs[k] = v; rq[k] = v * v;
  __syncthreads();
  for (int s = 128; s > 0; s >>= 1) {
    if (k < s) { rs[k] += rs[k + s]; rq[k] += rq[k + s]; }
    __syncthreads();
  }
  if (k == 0) {
    const float mean = rs[0] * (1.0f / E_);
    const float var  = rq[0] * (1.0f / E_) - mean * mean;
    stats[m] = make_float2(mean, rsqrtf(fmaxf(var, 0.0f) + EPS_));
  }
}

// ---------------------------------------------------------------------------
// Generic bf16 WMMA GEMM:  out[m,n] = epilogue( sum_k A(m,k) * W[n,k] )
// Block = 128 threads (4 waves). Tile: 16 rows x 64 cols, K-step 64
// (two WMMAs per barrier pair). A rows padded to 40 halves: lane offsets
// L*20 dwords mod 64 hit 16 disjoint 4-bank groups -> conflict-free b128.
// ---------------------------------------------------------------------------
#define APAD_ 40

struct GemmArgs {
  const unsigned short* W;      // bf16 weights, row-major [N][K]
  const float* bias;            // [N]
  int M, N, K;
  int amode;                    // 0 token(b=m/129,t=m%129), 1 token(b=m,t=0),
                                // 2 fp32 plain, 3 bf16 plain
  const float* Af32;
  const unsigned short* Abf;
  const float* xin;
  const unsigned short* obuf;
  const float2* stats;
  const float* ln1w;
  const float* ln2w;
  int emode;                    // 0 bf16 store, 1 f32 store, 2 f32 + residual,
                                // 3 exact-GELU -> bf16
  float* outf;
  unsigned short* outb;
  const float* res;
};

__device__ __forceinline__ float gemm_aval(const GemmArgs& a, int m, int kk) {
  switch (a.amode) {
    case 0: {
      const int b = m / TOK_, t = m % TOK_;
      const float2 st = a.stats[m];
      const float raw = (t == 0)
          ? a.xin[(size_t)b * E_ + kk]
          : bf2f(a.obuf[((size_t)b * H_ + (t - 1)) * E_ + kk]);
      const float lw = (t == 0) ? a.ln1w[kk] : a.ln2w[kk];
      return (raw - st.x) * st.y * lw;
    }
    case 1: {
      const float2 st = a.stats[(size_t)m * TOK_];
      return (a.xin[(size_t)m * E_ + kk] - st.x) * st.y * a.ln1w[kk];
    }
    case 2:  return a.Af32[(size_t)m * a.K + kk];
    default: return bf2f(a.Abf[(size_t)m * a.K + kk]);
  }
}

__global__ __launch_bounds__(128) void gemm_kernel(GemmArgs a) {
  __shared__ __align__(16) unsigned short As[2][16 * APAD_];  // 2 K-chunks
  const int tid  = threadIdx.x;
  const int wave = tid >> 5, lane = tid & 31;
  const int m0 = blockIdx.x * 16;
  const int n0 = blockIdx.y * 64 + wave * 16;

  v8f acc = {0.0f, 0.0f, 0.0f, 0.0f, 0.0f, 0.0f, 0.0f, 0.0f};

  const int row  = lane & 15;
  const int ro   = (lane & 16) ? 8 : 0;    // A K-subblock select per ISA layout
  const int ncol = n0 + (lane & 15);       // B column for this lane
  const int koff = (lane & 16) ? 8 : 0;    // B K-subblock select per ISA layout

  for (int k0 = 0; k0 < a.K; k0 += 64) {
    // stage + normalize two 16x32 A chunks
    for (int idx = tid; idx < 1024; idx += 128) {
      const int u  = idx >> 9;             // which 32-wide K chunk
      const int r2 = (idx >> 5) & 15, kk = idx & 31;
      As[u][r2 * APAD_ + kk] = f2bf(gemm_aval(a, m0 + r2, k0 + u * 32 + kk));
    }
    __syncthreads();

    #pragma unroll
    for (int u = 0; u < 2; ++u) {
      union { v16bf v; uint4 q[2]; } af, bfr;
      af.q[0] = *(const uint4*)&As[u][row * APAD_ + ro];
      af.q[1] = *(const uint4*)&As[u][row * APAD_ + ro + 16];
      const unsigned short* wp = a.W + (size_t)ncol * a.K + k0 + u * 32 + koff;
      bfr.q[0] = *(const uint4*)wp;
      bfr.q[1] = *(const uint4*)(wp + 16);
      __builtin_prefetch(wp + 64, 0, 0);   // next K-step of this weight row
      acc = __builtin_amdgcn_wmma_f32_16x16x32_bf16(
          false, af.v, false, bfr.v, (short)0, acc, false, false);
    }
    __syncthreads();
  }

  // epilogue: lane<16 holds rows 0..7, lane>=16 rows 8..15; col = lane&15
  const float bn = a.bias ? a.bias[ncol] : 0.0f;
  const int mb = (lane & 16) ? 8 : 0;
  #pragma unroll
  for (int v = 0; v < 8; ++v) {
    const int m = m0 + mb + v;
    float val = acc[v] + bn;
    const size_t o = (size_t)m * a.N + ncol;
    switch (a.emode) {
      case 0: a.outb[o] = f2bf(val); break;
      case 1: a.outf[o] = val; break;
      case 2: a.outf[o] = val + a.res[o]; break;
      default: {
        const float g = 0.5f * val * (1.0f + erff(val * 0.70710678118654752f));
        a.outb[o] = f2bf(g);
      } break;
    }
  }
}

// ---------------------------------------------------------------------------
// Kernel 4: attention. One block per batch row; 129-key softmax per head.
// ---------------------------------------------------------------------------
__global__ __launch_bounds__(256) void attn_kernel(
    const float* __restrict__ qws, const unsigned short* __restrict__ kbf,
    const unsigned short* __restrict__ vbf, float* __restrict__ ows) {
  const int b = blockIdx.x, tid = threadIdx.x;
  __shared__ float qs[E_], sc[256], red[256];
  qs[tid] = qws[(size_t)b * E_ + tid];
  __syncthreads();
  const float scale = 0.17677669529663687f;   // 1/sqrt(32)

  for (int h = 0; h < NH_; ++h) {
    float s = -1e30f;
    if (tid < TOK_) {
      const unsigned short* kp = kbf + ((size_t)b * TOK_ + tid) * E_ + h * HD_;
      float acc = 0.0f;
      #pragma unroll
      for (int d = 0; d < HD_; ++d) acc += qs[h * HD_ + d] * bf2f(kp[d]);
      s = acc * scale;
    }
    sc[tid] = s; red[tid] = s;
    __syncthreads();
    for (int st = 128; st > 0; st >>= 1) {
      if (tid < st) red[tid] = fmaxf(red[tid], red[tid + st]);
      __syncthreads();
    }
    const float mx = red[0];
    __syncthreads();
    const float p = (tid < TOK_) ? expf(sc[tid] - mx) : 0.0f;
    sc[tid] = p; red[tid] = p;
    __syncthreads();
    for (int st = 128; st > 0; st >>= 1) {
      if (tid < st) red[tid] += red[tid + st];
      __syncthreads();
    }
    const float inv = 1.0f / red[0];
    __syncthreads();
    if (tid < HD_) {
      float acc = 0.0f;
      for (int t = 0; t < TOK_; ++t)
        acc += sc[t] * bf2f(vbf[((size_t)b * TOK_ + t) * E_ + h * HD_ + tid]);
      ows[(size_t)b * E_ + h * HD_ + tid] = acc * inv;
    }
    __syncthreads();
  }
}

// ---------------------------------------------------------------------------
// Kernel 6: ln3 over residual rows -> bf16 y for the MLP.
// ---------------------------------------------------------------------------
__global__ __launch_bounds__(256) void ln3_kernel(
    const float* __restrict__ attnres, const float* __restrict__ w,
    unsigned short* __restrict__ ybf) {
  const int b = blockIdx.x, k = threadIdx.x;
  const float v = attnres[(size_t)b * E_ + k];
  __shared__ float rs[256], rq[256];
  rs[k] = v; rq[k] = v * v;
  __syncthreads();
  for (int s = 128; s > 0; s >>= 1) {
    if (k < s) { rs[k] += rs[k + s]; rq[k] += rq[k + s]; }
    __syncthreads();
  }
  const float mean = rs[0] * (1.0f / E_);
  const float var  = rq[0] * (1.0f / E_) - mean * mean;
  const float rstd = rsqrtf(fmaxf(var, 0.0f) + EPS_);
  ybf[(size_t)b * E_ + k] = f2bf((v - mean) * rstd * w[k]);
}

// ---------------------------------------------------------------------------
// Host launcher
// ---------------------------------------------------------------------------
extern "C" void kernel_launch(void* const* d_in, const int* in_sizes, int n_in,
                              void* d_out, int out_size, void* d_ws,
                              size_t ws_size, hipStream_t stream) {
  (void)in_sizes; (void)n_in; (void)out_size; (void)ws_size;
  const float* xin        = (const float*)d_in[0];
  const float* h0s        = (const float*)d_in[1];
  const float* Wih        = (const float*)d_in[2];
  const float* Whh        = (const float*)d_in[3];
  const float* bih        = (const float*)d_in[4];
  const float* bhh        = (const float*)d_in[5];
  const float* ln1w       = (const float*)d_in[6];
  const float* ln2w       = (const float*)d_in[7];
  const float* ln3w       = (const float*)d_in[8];
  const float* attn_in_w  = (const float*)d_in[9];
  const float* attn_in_b  = (const float*)d_in[10];
  const float* attn_out_w = (const float*)d_in[11];
  const float* attn_out_b = (const float*)d_in[12];
  const float* fc_w       = (const float*)d_in[13];
  const float* fc_b       = (const float*)d_in[14];
  const float* proj_w     = (const float*)d_in[15];
  const float* proj_b     = (const float*)d_in[16];

  float* out0 = (float*)d_out;                     // (B,E)
  float* hfin = out0 + (size_t)B_ * E_;            // (E,H)

  char* ws = (char*)d_ws;
  size_t cur = 0;
  auto alloc = [&](size_t bytes) -> void* {
    void* p = ws + cur;
    cur += (bytes + 255) & ~(size_t)255;
    return p;
  };
  unsigned short* obuf  = (unsigned short*)alloc((size_t)B_ * H_ * E_ * 2);
  float2*         stats = (float2*)alloc((size_t)B_ * TOK_ * sizeof(float2));
  float*          qws   = (float*)alloc((size_t)B_ * E_ * 4);
  unsigned short* kbf   = (unsigned short*)alloc((size_t)B_ * TOK_ * E_ * 2);
  unsigned short* vbf   = (unsigned short*)alloc((size_t)B_ * TOK_ * E_ * 2);
  float*          owsb  = (float*)alloc((size_t)B_ * E_ * 4);
  float*          ares  = (float*)alloc((size_t)B_ * E_ * 4);
  unsigned short* ybf   = (unsigned short*)alloc((size_t)B_ * E_ * 2);
  unsigned short* h1bf  = (unsigned short*)alloc((size_t)B_ * 4 * E_ * 2);
  unsigned short* wai   = (unsigned short*)alloc((size_t)3 * E_ * E_ * 2);
  unsigned short* wao   = (unsigned short*)alloc((size_t)E_ * E_ * 2);
  unsigned short* wfc   = (unsigned short*)alloc((size_t)4 * E_ * E_ * 2);
  unsigned short* wprj  = (unsigned short*)alloc((size_t)E_ * 4 * E_ * 2);
  uint4*          wbhh  = (uint4*)alloc((size_t)E_ * WTILE_ * sizeof(uint4));

  // weight conversions (tiny; L2-resident afterwards)
  cvt_kernel<<<(3 * E_ * E_ + 255) / 256, 256, 0, stream>>>(attn_in_w, wai, 3 * E_ * E_);
  cvt_kernel<<<(E_ * E_ + 255) / 256, 256, 0, stream>>>(attn_out_w, wao, E_ * E_);
  cvt_kernel<<<(4 * E_ * E_ + 255) / 256, 256, 0, stream>>>(fc_w, wfc, 4 * E_ * E_);
  cvt_kernel<<<(E_ * 4 * E_ + 255) / 256, 256, 0, stream>>>(proj_w, wprj, E_ * 4 * E_);
#if HAVE_ASYNC_LDS
  cvt_whh_kernel<<<E_, 128, 0, stream>>>(Whh, wbhh);
#endif

  // 1) per-channel GRU scan (LDS-resident weights via async DMA)
  gru_kernel<<<E_, 128, 0, stream>>>(xin, h0s, Wih, Whh, bih, bhh, wbhh,
                                     obuf, hfin);

  // 2) token layernorm stats
  stats_kernel<<<B_ * TOK_, 256, 0, stream>>>(xin, obuf, stats);

  GemmArgs g{};
  g.xin = xin; g.obuf = obuf; g.stats = stats; g.ln1w = ln1w; g.ln2w = ln2w;

  // 3) q = ln1(x) @ Wq^T + bq  (fp32)
  g.W = wai; g.bias = attn_in_b; g.M = B_; g.N = E_; g.K = E_;
  g.amode = 1; g.emode = 1; g.outf = qws;
  gemm_kernel<<<dim3(B_ / 16, E_ / 64), 128, 0, stream>>>(g);

  // 4) k = kv @ Wk^T + bk  (bf16)
  g.W = wai + (size_t)E_ * E_; g.bias = attn_in_b + E_;
  g.M = B_ * TOK_; g.amode = 0; g.emode = 0; g.outb = kbf; g.outf = nullptr;
  gemm_kernel<<<dim3(B_ * TOK_ / 16, E_ / 64), 128, 0, stream>>>(g);

  // 5) v = kv @ Wv^T + bv  (bf16)
  g.W = wai + (size_t)2 * E_ * E_; g.bias = attn_in_b + 2 * E_; g.outb = vbf;
  gemm_kernel<<<dim3(B_ * TOK_ / 16, E_ / 64), 128, 0, stream>>>(g);

  // 6) softmax attention
  attn_kernel<<<B_, 256, 0, stream>>>(qws, kbf, vbf, owsb);

  // 7) attn_out = o @ Wo^T + bo + x   (residual base)
  GemmArgs g2{};
  g2.W = wao; g2.bias = attn_out_b; g2.M = B_; g2.N = E_; g2.K = E_;
  g2.amode = 2; g2.Af32 = owsb; g2.emode = 2; g2.outf = ares; g2.res = xin;
  gemm_kernel<<<dim3(B_ / 16, E_ / 64), 128, 0, stream>>>(g2);

  // 8) y = ln3(attn_out)
  ln3_kernel<<<B_, 256, 0, stream>>>(ares, ln3w, ybf);

  // 9) h1 = gelu(y @ fc_w^T + fc_b)
  GemmArgs g3{};
  g3.W = wfc; g3.bias = fc_b; g3.M = B_; g3.N = 4 * E_; g3.K = E_;
  g3.amode = 3; g3.Abf = ybf; g3.emode = 3; g3.outb = h1bf;
  gemm_kernel<<<dim3(B_ / 16, (4 * E_) / 64), 128, 0, stream>>>(g3);

  // 10) out = attn_out + h1 @ proj_w^T + proj_b
  GemmArgs g4{};
  g4.W = wprj; g4.bias = proj_b; g4.M = B_; g4.N = E_; g4.K = 4 * E_;
  g4.amode = 3; g4.Abf = h1bf; g4.emode = 2; g4.outf = out0; g4.res = ares;
  gemm_kernel<<<dim3(B_ / 16, E_ / 64), 128, 0, stream>>>(g4);
}
